// SelfAttention_84000970375534
// MI455X (gfx1250) — matrix operand order repaired and checked by reading, not compiled, not run on hardware
//
#include <hip/hip_runtime.h>

// ---------------------------------------------------------------------------
// Multi-head self-attention forward for MI455X (gfx1250, wave32, WMMA).
//   B=2, S=2048, D=1024, H=16, Dh=64.
// bf16 operands + fp32 WMMA accumulation (v_wmma_f32_16x16x32_bf16),
// flash-attention fusion. Weights and V are PRE-PACKED into the exact WMMA
// B-fragment panel layout so every hot-loop load is per-lane contiguous
// (lowers to global_load_b128), eliminating strided u16 loads + v_mov_b16
// repacking seen in the previous round's histogram.
//
// B-fragment panel layout (32Kx16N tile -> 512 bf16):
//   element (k,n) of panel -> offset ( ((k>>4)*16 + (n&15))*16 + (k&15) )
//   i.e. lane L = (k>=16)*16 + n, elements e=0..15 contiguous with k = kb+e.
// ---------------------------------------------------------------------------

#define BATCH 2
#define SEQ   2048
#define DM    1024
#define HEADS 16
#define DH    64
#define KP    (DM / 32)   // 32 k-panels per column group

typedef __bf16 bf16_t;
typedef __attribute__((ext_vector_type(16))) __bf16 v16bf;
typedef __attribute__((ext_vector_type(8)))  float  v8f;

// A fragment (16x32 bf16): lane m = lane&15; e -> k = (lane>=16?16:0) + e
// C/D (16x16 f32): vgpr r -> m = (lane>=16?8:0)+r ; n = lane&15

__device__ __forceinline__ v16bf load_a_frag(const bf16_t* __restrict__ rowptr,
                                             int khalf8) {
  v16bf a;
#pragma unroll
  for (int e = 0; e < 16; ++e)
    a[e] = rowptr[(khalf8 << 1) + e];
  return a;
}

__device__ __forceinline__ v16bf load_b_frag_contig(const bf16_t* __restrict__ p) {
  v16bf b;
#pragma unroll
  for (int e = 0; e < 16; ++e) b[e] = p[e];
  return b;
}

__device__ __forceinline__ v8f wmma_bf16(v16bf a, v16bf b, v8f c) {
  return __builtin_amdgcn_wmma_f32_16x16x32_bf16(false, a, false, b,
                                                 (short)0, c, false, false);
}

// ------------------------- kernel 0a: fp32 -> bf16 -------------------------
__global__ void cvt_f32_to_bf16(const float* __restrict__ src,
                                bf16_t* __restrict__ dst, int n) {
  int i = blockIdx.x * blockDim.x + threadIdx.x;
  if (i < n) dst[i] = (bf16_t)src[i];
}

// -------------- kernel 0b: pack weight [DM,DM] f32 -> B-frag panels --------
// panel index = (n>>4)*KP + (k>>5); within panel: ((kmid*16 + n16)*16 + klo)
__global__ void pack_weight_bfrag(const float* __restrict__ W,
                                  bf16_t* __restrict__ dst) {
  int i = blockIdx.x * blockDim.x + threadIdx.x;
  if (i >= DM * DM) return;
  const int k = i >> 10;        // row (K dim)
  const int n = i & (DM - 1);   // col (N dim)
  const int panel = (n >> 4) * KP + (k >> 5);
  const int pos = ((((k >> 4) & 1) * 16 + (n & 15)) << 4) + (k & 15);
  dst[(size_t)panel * 512 + pos] = (bf16_t)W[i];
}

// ---------------- kernel 1: projection GEMM (packed weights) ---------------
// X: [B*S, DM] bf16 row-major. Wpk: packed B panels. Wave tile: 32(M) x 64(N).
// vmode 0: out = head-split [bh][s][dh]   (Q and K)
// vmode 1: out = V packed B panels per bh: [bh][s/32][dhchunk(4)][512]
__global__ __launch_bounds__(128)
void gemm_proj(const bf16_t* __restrict__ X, const bf16_t* __restrict__ Wpk,
               const float* __restrict__ bias, bf16_t* __restrict__ out,
               int vmode) {
  const int lane   = threadIdx.x & 31;
  const int wave   = threadIdx.x >> 5;
  const int mtile  = blockIdx.x;                    // 32 rows of M=4096
  const int nbase  = (blockIdx.y * 4 + wave) * 64;  // 64 cols of N=1024
  const int n16    = lane & 15;
  const int khalf8 = (lane >> 4) * 8;

  v8f acc0[4] = {}, acc1[4] = {};

  const bf16_t* Xrow0 = X + (size_t)(mtile * 32 + n16) * DM;
  const bf16_t* Xrow1 = Xrow0 + (size_t)16 * DM;
  const bf16_t* Wbase = Wpk + (size_t)(nbase >> 4) * KP * 512 + lane * 16;

  for (int kp = 0; kp < KP; ++kp) {
    v16bf a0 = load_a_frag(Xrow0 + kp * 32, khalf8);
    v16bf a1 = load_a_frag(Xrow1 + kp * 32, khalf8);
#pragma unroll
    for (int c = 0; c < 4; ++c) {
      v16bf b = load_b_frag_contig(Wbase + ((size_t)c * KP + kp) * 512);
      acc0[c] = wmma_bf16(a0, b, acc0[c]);
      acc1[c] = wmma_bf16(a1, b, acc1[c]);
    }
  }

#pragma unroll
  for (int c = 0; c < 4; ++c) {
    const int n  = nbase + c * 16 + n16;
    const int h  = n >> 6;
    const int dh = n & (DH - 1);
    const float bn = bias[n];
#pragma unroll
    for (int half = 0; half < 2; ++half) {
      const v8f& acc = half ? acc1[c] : acc0[c];
#pragma unroll
      for (int r = 0; r < 8; ++r) {
        const int gm = mtile * 32 + half * 16 + khalf8 + r; // row in [0,B*S)
        const int bb = gm >> 11;
        const int ss = gm & (SEQ - 1);
        const int bh = bb * HEADS + h;
        const float v = acc[r] + bn;
        if (vmode == 0) {
          out[((size_t)bh * SEQ + ss) * DH + dh] = (bf16_t)v;
        } else {
          // V packed: panel = ss>>5, chunk = dh>>4,
          // pos = ((kmid*16 + (dh&15))*16 + klo), key-in-panel = ss&31
          const int pos = ((((ss >> 4) & 1) * 16 + (dh & 15)) << 4) + (ss & 15);
          out[((size_t)bh * (SEQ / 32) + (ss >> 5)) * 2048 + (dh >> 4) * 512 + pos] =
              (bf16_t)v;
        }
      }
    }
  }
}

// ---------------------- kernel 2: flash attention --------------------------
// Q,K: [bh][S][DH] bf16.  Vpk: packed B panels.  mask: [B][S].  Y: [B][S][DM] bf16.
__global__ __launch_bounds__(128)
void flash_attn(const bf16_t* __restrict__ Q, const bf16_t* __restrict__ K,
                const bf16_t* __restrict__ Vpk, const int* __restrict__ mask,
                bf16_t* __restrict__ Y) {
  __shared__ float ldsP[4][16 * 32];   // per-wave P re-layout buffer

  const int lane   = threadIdx.x & 31;
  const int wave   = threadIdx.x >> 5;
  const int n16    = lane & 15;
  const int khalf8 = (lane >> 4) * 8;
  const int kb16   = (lane >> 4) * 16;

  const int tile   = blockIdx.x * 4 + wave;
  const int qtiles = SEQ / 16;
  const int bh     = tile / qtiles;
  const int qb     = (tile % qtiles) * 16;
  const int b      = bh / HEADS;
  const int h      = bh - b * HEADS;
  const float scale = 0.125f;          // 1/sqrt(DH)

  const bf16_t* Qrow = Q + ((size_t)bh * SEQ + qb + n16) * DH;
  v16bf aq0 = load_a_frag(Qrow, khalf8);
  v16bf aq1 = load_a_frag(Qrow + 32, khalf8);

  v8f o[4] = {};
  float mrow[8], lrow[8];
#pragma unroll
  for (int r = 0; r < 8; ++r) { mrow[r] = -3.0e38f; lrow[r] = 0.0f; }

  const int* maskb = mask + b * SEQ;
  float* P = ldsP[wave];

  for (int j = 0; j < SEQ; j += 32) {
    // ---- logits (K loads contiguous per lane: dh-major) ----
    const bf16_t* Kr1 = K + ((size_t)bh * SEQ + j + n16) * DH + kb16;
    const bf16_t* Kr2 = K + ((size_t)bh * SEQ + j + 16 + n16) * DH + kb16;
    v8f s1 = {}, s2 = {};
    s1 = wmma_bf16(aq0, load_b_frag_contig(Kr1), s1);
    s1 = wmma_bf16(aq1, load_b_frag_contig(Kr1 + 32), s1);
    s2 = wmma_bf16(aq0, load_b_frag_contig(Kr2), s2);
    s2 = wmma_bf16(aq1, load_b_frag_contig(Kr2 + 32), s2);

    // ---- masked online softmax ----
    const int mv1 = maskb[j + n16];
    const int mv2 = maskb[j + 16 + n16];
#pragma unroll
    for (int r = 0; r < 8; ++r) {
      float x1 = mv1 ? s1[r] * scale : -3.0e38f;
      float x2 = mv2 ? s2[r] * scale : -3.0e38f;
      float tmax = fmaxf(x1, x2);
#pragma unroll
      for (int off = 8; off >= 1; off >>= 1)
        tmax = fmaxf(tmax, __shfl_xor(tmax, off, 32));
      const float mnew  = fmaxf(mrow[r], tmax);
      const float alpha = __expf(mrow[r] - mnew);
      const float p1 = mv1 ? __expf(x1 - mnew) : 0.0f;
      const float p2 = mv2 ? __expf(x2 - mnew) : 0.0f;
      float ts = p1 + p2;
#pragma unroll
      for (int off = 8; off >= 1; off >>= 1)
        ts += __shfl_xor(ts, off, 32);
      lrow[r] = lrow[r] * alpha + ts;
      mrow[r] = mnew;
#pragma unroll
      for (int c = 0; c < 4; ++c) o[c][r] *= alpha;
      P[(khalf8 + r) * 32 + n16]      = p1;
      P[(khalf8 + r) * 32 + 16 + n16] = p2;
    }

    // ---- O += P * V  (V fragments contiguous: packed panels) ----
    v16bf ap;
#pragma unroll
    for (int e = 0; e < 16; ++e)
      ap[e] = (bf16_t)P[n16 * 32 + (khalf8 << 1) + e];
    const bf16_t* Vp = Vpk + ((size_t)bh * (SEQ / 32) + (j >> 5)) * 2048 + lane * 16;
#pragma unroll
    for (int c = 0; c < 4; ++c) {
      v16bf bv = load_b_frag_contig(Vp + c * 512);
      o[c] = wmma_bf16(ap, bv, o[c]);
    }
  }

  // ---- normalize and store Y [B][S][DM] bf16 (merge heads) ----
#pragma unroll
  for (int r = 0; r < 8; ++r) {
    const float inv = (lrow[r] > 0.0f) ? 1.0f / lrow[r] : 0.0f; // all-masked -> 0
    const int ss = qb + khalf8 + r;
#pragma unroll
    for (int c = 0; c < 4; ++c)
      Y[((size_t)b * SEQ + ss) * DM + h * DH + c * 16 + n16] =
          (bf16_t)(o[c][r] * inv);
  }
}

// ---------------- kernel 3: output projection (packed Wp) ------------------
__global__ __launch_bounds__(128)
void gemm_out(const bf16_t* __restrict__ Yb, const bf16_t* __restrict__ Wpk,
              const float* __restrict__ bias, float* __restrict__ out) {
  const int lane   = threadIdx.x & 31;
  const int wave   = threadIdx.x >> 5;
  const int mtile  = blockIdx.x;
  const int nbase  = (blockIdx.y * 4 + wave) * 64;
  const int n16    = lane & 15;
  const int khalf8 = (lane >> 4) * 8;

  v8f acc0[4] = {}, acc1[4] = {};
  const bf16_t* Yrow0 = Yb + (size_t)(mtile * 32 + n16) * DM;
  const bf16_t* Yrow1 = Yrow0 + (size_t)16 * DM;
  const bf16_t* Wbase = Wpk + (size_t)(nbase >> 4) * KP * 512 + lane * 16;

  for (int kp = 0; kp < KP; ++kp) {
    v16bf a0 = load_a_frag(Yrow0 + kp * 32, khalf8);
    v16bf a1 = load_a_frag(Yrow1 + kp * 32, khalf8);
#pragma unroll
    for (int c = 0; c < 4; ++c) {
      v16bf b = load_b_frag_contig(Wbase + ((size_t)c * KP + kp) * 512);
      acc0[c] = wmma_bf16(a0, b, acc0[c]);
      acc1[c] = wmma_bf16(a1, b, acc1[c]);
    }
  }

#pragma unroll
  for (int c = 0; c < 4; ++c) {
    const int n = nbase + c * 16 + n16;
    const float bn = bias[n];
#pragma unroll
    for (int r = 0; r < 8; ++r) {
      const int gm = mtile * 32 + khalf8 + r;
      out[(size_t)gm * DM + n]        = acc0[c][r] + bn;
      out[(size_t)(gm + 16) * DM + n] = acc1[c][r] + bn;
    }
  }
}

// ------------------------------- launcher ----------------------------------
extern "C" void kernel_launch(void* const* d_in, const int* in_sizes, int n_in,
                              void* d_out, int out_size, void* d_ws, size_t ws_size,
                              hipStream_t stream) {
  (void)in_sizes; (void)n_in; (void)out_size; (void)ws_size;

  const float* x    = (const float*)d_in[0];
  const int*   mask = (const int*)  d_in[1];
  const float* Wq   = (const float*)d_in[2];
  const float* bq   = (const float*)d_in[3];
  const float* Wk   = (const float*)d_in[4];
  const float* bk   = (const float*)d_in[5];
  const float* Wv   = (const float*)d_in[6];
  const float* bv   = (const float*)d_in[7];
  const float* Wp   = (const float*)d_in[8];
  const float* bp   = (const float*)d_in[9];
  float* out = (float*)d_out;

  const int NX = BATCH * SEQ * DM;
  const int NW = DM * DM;

  char* ws = (char*)d_ws;
  size_t off = 0;
  bf16_t* xb  = (bf16_t*)(ws + off); off += (size_t)NX * 2;
  bf16_t* Wqp = (bf16_t*)(ws + off); off += (size_t)NW * 2;
  bf16_t* Wkp = (bf16_t*)(ws + off); off += (size_t)NW * 2;
  bf16_t* Wvp = (bf16_t*)(ws + off); off += (size_t)NW * 2;
  bf16_t* Wpp = (bf16_t*)(ws + off); off += (size_t)NW * 2;
  bf16_t* Qb  = (bf16_t*)(ws + off); off += (size_t)NX * 2;
  bf16_t* Kb  = (bf16_t*)(ws + off); off += (size_t)NX * 2;
  bf16_t* Vpk = (bf16_t*)(ws + off); off += (size_t)NX * 2;
  bf16_t* Yb  = (bf16_t*)(ws + off); off += (size_t)NX * 2;

  // 0) convert x; pack weights into B-fragment panels
  cvt_f32_to_bf16<<<(NX + 255) / 256, 256, 0, stream>>>(x, xb, NX);
  pack_weight_bfrag<<<(NW + 255) / 256, 256, 0, stream>>>(Wq, Wqp);
  pack_weight_bfrag<<<(NW + 255) / 256, 256, 0, stream>>>(Wk, Wkp);
  pack_weight_bfrag<<<(NW + 255) / 256, 256, 0, stream>>>(Wv, Wvp);
  pack_weight_bfrag<<<(NW + 255) / 256, 256, 0, stream>>>(Wp, Wpp);

  // 1) Q/K/V projections: M=4096 -> 128 m-tiles of 32; N=1024 -> 4x4x64
  dim3 gGemm(128, 4, 1), bGemm(128, 1, 1);
  gemm_proj<<<gGemm, bGemm, 0, stream>>>(xb, Wqp, bq, Qb, 0);
  gemm_proj<<<gGemm, bGemm, 0, stream>>>(xb, Wkp, bk, Kb, 0);
  gemm_proj<<<gGemm, bGemm, 0, stream>>>(xb, Wvp, bv, Vpk, 1);

  // 2) flash attention: 4096 q-tiles, 4 waves per block
  flash_attn<<<1024, 128, 0, stream>>>(Qb, Kb, Vpk, mask, Yb);

  // 3) output projection -> fp32
  gemm_out<<<gGemm, bGemm, 0, stream>>>(Yb, Wpp, bp, out);
}
